// LoopGPT_36137854828697
// MI455X (gfx1250) — matrix operand order repaired and checked by reading, not compiled, not run on hardware
//
#include <hip/hip_runtime.h>
#include <cstdint>
#include <cstddef>

// ---------------- CDNA5 (gfx1250) wave32 WMMA types ----------------
typedef __attribute__((ext_vector_type(16))) __bf16 v16bf;
typedef __attribute__((ext_vector_type(8)))  __bf16 v8bf;
typedef __attribute__((ext_vector_type(8)))  float  v8f;

namespace {
constexpr int Vv = 50257;   // vocab
constexpr int Tt = 1024;    // seq len
constexpr int Nn = 768;     // embed
constexpr int Hh = 12;      // heads
constexpr int Ll = 12;      // loops
constexpr int Bb = 8;       // batch
constexpr int Mrows = Bb * Tt;   // 8192 token rows
constexpr int N3 = 3 * Nn;       // 2304
constexpr int N4 = 4 * Nn;       // 3072
}

__device__ __forceinline__ v8f zero8() {
  v8f z;
#pragma unroll
  for (int i = 0; i < 8; ++i) z[i] = 0.f;
  return z;
}

__device__ __forceinline__ v16bf cat8(v8bf lo, v8bf hi) {
  v16bf r;
#pragma unroll
  for (int i = 0; i < 8; ++i) { r[i] = lo[i]; r[i + 8] = hi[i]; }
  return r;
}

__device__ __forceinline__ v8f wmma_bf16(v16bf a, v16bf b, v8f c) {
  // D = A(16x32 bf16) * B(32x16 bf16) + C(16x16 f32)
  return __builtin_amdgcn_wmma_f32_16x16x32_bf16(false, a, false, b, (short)0, c,
                                                 false, false);
}

// ---------------- elementwise f32 -> bf16 ----------------
__global__ void k_f2bf(const float* __restrict__ src, __bf16* __restrict__ dst,
                       long long n) {
  long long i = (long long)blockIdx.x * blockDim.x + threadIdx.x;
  if (i < n) dst[i] = (__bf16)src[i];
}

// ---------------- pack f32 weight [K,Nc] into WMMA B-fragment order ----------------
// Fragment element i of lane L (half = L>>4) covers k = ks*32 + half*16 + i, n = nt*16 + (L&15).
// Packed index = ((nt*Ksteps + ks)*32 + lane)*16 + i  -> 32B contiguous per lane.
__global__ void k_pack_w(const float* __restrict__ W, __bf16* __restrict__ P,
                         int K, int Nc) {
  int idx = blockIdx.x * blockDim.x + threadIdx.x;
  int ksteps = K >> 5;
  int total = (Nc >> 4) * ksteps * 512;
  if (idx >= total) return;
  int i = idx & 15;
  int lane = (idx >> 4) & 31;
  int rest = idx >> 9;
  int ks = rest % ksteps;
  int nt = rest / ksteps;
  int hl = lane >> 4;
  int n = nt * 16 + (lane & 15);
  int k = ks * 32 + hl * 16 + i;
  P[idx] = (__bf16)W[(size_t)k * Nc + n];
}

// ---------------- token + position embedding; h := x0 ----------------
__global__ void k_embed(const int* __restrict__ idx, const float* __restrict__ wte,
                        const float* __restrict__ wpe, float* __restrict__ x0,
                        float* __restrict__ h) {
  int i = blockIdx.x * blockDim.x + threadIdx.x;   // over Mrows*Nn
  int n = i % Nn;
  int row = i / Nn;
  int t = row % Tt;
  float v = wte[(size_t)idx[row] * Nn + n] + wpe[(size_t)t * Nn + n];
  x0[i] = v;
  h[i] = v;
}

// ---------------- fused (A [+B] [+C]) add + LayerNorm -> bf16 (optional f32 copy) ----------------
__global__ void k_add_ln(const float* __restrict__ A, const float* __restrict__ Bp,
                         const float* __restrict__ Cp, const float* __restrict__ g,
                         const float* __restrict__ beta, float* __restrict__ x_out,
                         __bf16* __restrict__ ln_out) {
  int row = blockIdx.x;           // one row of 768 per block
  int tid = threadIdx.x;          // 256 threads, 3 elems each
  __shared__ float red[256];
  float vals[3];
  float s = 0.f;
#pragma unroll
  for (int e = 0; e < 3; ++e) {
    int n = tid + e * 256;
    float v = A[(size_t)row * Nn + n];
    if (Bp) v += Bp[(size_t)row * Nn + n];
    if (Cp) v += Cp[n];
    vals[e] = v;
    s += v;
    if (x_out) x_out[(size_t)row * Nn + n] = v;
  }
  red[tid] = s;
  __syncthreads();
  for (int o = 128; o > 0; o >>= 1) { if (tid < o) red[tid] += red[tid + o]; __syncthreads(); }
  float mean = red[0] * (1.f / Nn);
  __syncthreads();
  float s2 = 0.f;
#pragma unroll
  for (int e = 0; e < 3; ++e) { float d = vals[e] - mean; s2 += d * d; }
  red[tid] = s2;
  __syncthreads();
  for (int o = 128; o > 0; o >>= 1) { if (tid < o) red[tid] += red[tid + o]; __syncthreads(); }
  float rstd = rsqrtf(red[0] * (1.f / Nn) + 1e-5f);
#pragma unroll
  for (int e = 0; e < 3; ++e) {
    int n = tid + e * 256;
    ln_out[(size_t)row * Nn + n] = (__bf16)((vals[e] - mean) * rstd * g[n] + beta[n]);
  }
}

// ---------------- bf16 GEMM: wave computes 16(M) x 64(N) tile ----------------
// A: bf16 row-major [Mrows,K]; Bp: packed B-fragments; modes:
//   0: out_bf = bf16(acc + bias)
//   1: out_f  = acc + bias + resid
//   2: out_bf = bf16(gelu_exact(acc + bias))
__global__ void k_gemm(const __bf16* __restrict__ A, const __bf16* __restrict__ Bp,
                       const float* __restrict__ bias, const float* __restrict__ resid,
                       float* __restrict__ outf, __bf16* __restrict__ outh,
                       int Mr, int K, int Nc, int tiles_n4, int mode) {
  int wave = blockIdx.x * (blockDim.x >> 5) + (threadIdx.x >> 5);
  int tiles_m = Mr >> 4;
  if (wave >= tiles_m * tiles_n4) return;
  int lane = threadIdx.x & 31;
  int hl = lane >> 4;
  int l16 = lane & 15;
  int tm = wave / tiles_n4;
  int tn4 = wave % tiles_n4;
  int ksteps = K >> 5;

  v8f acc[4];
#pragma unroll
  for (int t = 0; t < 4; ++t) acc[t] = zero8();

  const __bf16* arow = A + (size_t)(tm * 16 + l16) * K + hl * 16;
  for (int ks = 0; ks < ksteps; ++ks) {
    v8bf alo = *(const v8bf*)(arow + ks * 32);
    v8bf ahi = *(const v8bf*)(arow + ks * 32 + 8);
    v16bf a = cat8(alo, ahi);
#pragma unroll
    for (int t = 0; t < 4; ++t) {
      const __bf16* bp =
          Bp + ((((size_t)(tn4 * 4 + t) * ksteps + ks) * 32 + lane) << 4);
      v16bf b = *(const v16bf*)bp;
      acc[t] = wmma_bf16(a, b, acc[t]);
    }
  }

#pragma unroll
  for (int t = 0; t < 4; ++t) {
    int n = (tn4 * 4 + t) * 16 + l16;
    float bv = bias ? bias[n] : 0.f;
#pragma unroll
    for (int j = 0; j < 8; ++j) {
      int m = tm * 16 + hl * 8 + j;          // C layout: row = j + 8*half
      float v = acc[t][j] + bv;
      size_t off = (size_t)m * Nc + n;
      if (mode == 0) {
        outh[off] = (__bf16)v;
      } else if (mode == 1) {
        outf[off] = v + resid[off];
      } else {
        float ge = 0.5f * v * (1.f + erff(v * 0.70710678118654752f));
        outh[off] = (__bf16)ge;
      }
    }
  }
}

// ---------------- flash attention: one wave per (batch, head, 16-query tile) ----------------
__global__ void k_attn(const __bf16* __restrict__ qkv, __bf16* __restrict__ attn_o) {
  int bid = blockIdx.x;                 // B*H*(T/16) = 6144
  int qt = bid & 63;
  int hh = (bid >> 6) % Hh;
  int b = bid / (64 * Hh);
  int lane = threadIdx.x & 31;
  int hl = lane >> 4;
  int l16 = lane & 15;
  int q0 = qt * 16;
  size_t rowbase = (size_t)b * Tt * N3;

  // Q A-fragments (held in regs): element i -> d = hl*16 + i (+32 for qa1)
  const __bf16* qp = qkv + rowbase + (size_t)(q0 + l16) * N3 + hh * 64;
  v16bf qa0 = cat8(*(const v8bf*)(qp + hl * 16), *(const v8bf*)(qp + 8 + hl * 16));
  v16bf qa1 = cat8(*(const v8bf*)(qp + 32 + hl * 16), *(const v8bf*)(qp + 40 + hl * 16));

  __shared__ __bf16 Pb[16 * 32];   // P tile, row-major [q][key]
  __shared__ __bf16 Vt[64 * 32];   // transposed V chunk [d][key]

  v8f oacc[4];
#pragma unroll
  for (int t = 0; t < 4; ++t) oacc[t] = zero8();
  float rm[8], rl[8];
#pragma unroll
  for (int j = 0; j < 8; ++j) { rm[j] = -1e30f; rl[j] = 0.f; }

  int nchunk = (q0 + 16 + 31) >> 5;
  for (int c = 0; c < nchunk; ++c) {
    int kbase = c * 32;
    // ---- S = Q K^T for two 16-key tiles ----
    v8f s0 = zero8(), s1 = zero8();
#pragma unroll
    for (int ts = 0; ts < 2; ++ts) {
      const __bf16* kp =
          qkv + rowbase + (size_t)(kbase + ts * 16 + l16) * N3 + Nn + hh * 64;
      v16bf kb0 = *(const v16bf*)(kp + hl * 16);        // B-frag: d = hl*16 + i
      v16bf kb1 = *(const v16bf*)(kp + 32 + hl * 16);   // d = 32 + hl*16 + i
      v8f s = zero8();
      s = wmma_bf16(qa0, kb0, s);
      s = wmma_bf16(qa1, kb1, s);
      if (ts == 0) s0 = s; else s1 = s;
    }
    // ---- scale + causal mask (row q = q0 + j + 8*hl, col key = lane%16) ----
#pragma unroll
    for (int j = 0; j < 8; ++j) {
      int q = q0 + j + 8 * hl;
      int k0 = kbase + l16;
      int k1 = kbase + 16 + l16;
      s0[j] = (k0 <= q) ? s0[j] * 0.125f : -1e30f;
      s1[j] = (k1 <= q) ? s1[j] * 0.125f : -1e30f;
    }
    // ---- online softmax (reduce across 16 lanes within each half) ----
    float mnew[8], corr[8];
#pragma unroll
    for (int j = 0; j < 8; ++j) {
      float t = fmaxf(s0[j], s1[j]);
      for (int o = 8; o >= 1; o >>= 1) t = fmaxf(t, __shfl_xor(t, o, 16));
      mnew[j] = fmaxf(rm[j], t);
      corr[j] = __expf(rm[j] - mnew[j]);
    }
#pragma unroll
    for (int j = 0; j < 8; ++j) {
      s0[j] = __expf(s0[j] - mnew[j]);
      s1[j] = __expf(s1[j] - mnew[j]);
    }
#pragma unroll
    for (int j = 0; j < 8; ++j) {
      float t = s0[j] + s1[j];
      for (int o = 8; o >= 1; o >>= 1) t += __shfl_xor(t, o, 16);
      rl[j] = rl[j] * corr[j] + t;
      rm[j] = mnew[j];
    }
#pragma unroll
    for (int t = 0; t < 4; ++t)
#pragma unroll
      for (int j = 0; j < 8; ++j) oacc[t][j] *= corr[j];

    // ---- P -> LDS (transpose to A-frag order) ----
    __syncthreads();  // single-wave WG: S_NOP, ordering via DScnt
#pragma unroll
    for (int j = 0; j < 8; ++j) {
      Pb[(j + 8 * hl) * 32 + l16] = (__bf16)s0[j];
      Pb[(j + 8 * hl) * 32 + 16 + l16] = (__bf16)s1[j];
    }
    // ---- stage V chunk transposed: lane = key, 64 d values ----
    {
      const __bf16* vp = qkv + rowbase + (size_t)(kbase + lane) * N3 + 2 * Nn + hh * 64;
#pragma unroll
      for (int dd = 0; dd < 64; dd += 8) {
        v8bf vv = *(const v8bf*)(vp + dd);
#pragma unroll
        for (int e = 0; e < 8; ++e) Vt[(dd + e) * 32 + lane] = vv[e];
      }
    }
    __syncthreads();
    // ---- O += P V ----
    v16bf pa = cat8(*(const v8bf*)(Pb + l16 * 32 + hl * 16),
                    *(const v8bf*)(Pb + l16 * 32 + 8 + hl * 16));
#pragma unroll
    for (int t = 0; t < 4; ++t) {
      v16bf vb = *(const v16bf*)(Vt + (t * 16 + l16) * 32 + hl * 16);
      oacc[t] = wmma_bf16(pa, vb, oacc[t]);
    }
  }

  // ---- normalize, write bf16 attention output ----
#pragma unroll
  for (int t = 0; t < 4; ++t) {
    int col = hh * 64 + t * 16 + l16;
#pragma unroll
    for (int j = 0; j < 8; ++j) {
      int row = q0 + j + 8 * hl;
      attn_o[((size_t)b * Tt + row) * Nn + col] = (__bf16)(oacc[t][j] / rl[j]);
    }
  }
}

// ---------------- lm head: logits[b, v] = hf[last] . wte[v] (tied) ----------------
__global__ void k_lmhead(const __bf16* __restrict__ hfb, const __bf16* __restrict__ wteb,
                         float* __restrict__ logits) {
  constexpr int VT = (Vv + 15) / 16;   // 3142 vocab tiles
  int wave = blockIdx.x * (blockDim.x >> 5) + (threadIdx.x >> 5);
  if (wave >= VT) return;
  int lane = threadIdx.x & 31;
  int hl = lane >> 4;
  int l16 = lane & 15;
  int mrow = (l16 < Bb) ? l16 : 0;     // rows 8..15 compute garbage, never stored
  const __bf16* ap = hfb + ((size_t)mrow * Tt + (Tt - 1)) * Nn + hl * 16;
  int v = wave * 16 + l16;
  int vc = (v < Vv) ? v : (Vv - 1);
  const __bf16* bp = wteb + (size_t)vc * Nn + hl * 16;   // wte^T as B: k = n contiguous
  v8f acc = zero8();
  for (int ks = 0; ks < (Nn >> 5); ++ks) {
    v16bf a = cat8(*(const v8bf*)(ap + ks * 32), *(const v8bf*)(ap + ks * 32 + 8));
    v16bf b = *(const v16bf*)(bp + ks * 32);
    acc = wmma_bf16(a, b, acc);
  }
  if (hl == 0 && v < Vv) {
#pragma unroll
    for (int j = 0; j < 8; ++j) logits[(size_t)j * Vv + v] = acc[j];   // row m = j = batch
  }
}

// ---------------- steps_taken = L everywhere ----------------
__global__ void k_steps(int* __restrict__ s) {
  int i = blockIdx.x * blockDim.x + threadIdx.x;
  if (i < Bb * Tt) s[i] = Ll;
}

// =====================================================================
extern "C" void kernel_launch(void* const* d_in, const int* in_sizes, int n_in,
                              void* d_out, int out_size, void* d_ws, size_t ws_size,
                              hipStream_t stream) {
  (void)in_sizes; (void)n_in; (void)out_size; (void)ws_size;
  const int*   idx    = (const int*)  d_in[0];
  const float* wte    = (const float*)d_in[1];
  const float* wpe    = (const float*)d_in[2];
  const float* se     = (const float*)d_in[3];
  const float* ln1_g  = (const float*)d_in[4];
  const float* ln1_b  = (const float*)d_in[5];
  const float* attn_w = (const float*)d_in[6];
  const float* attn_b = (const float*)d_in[7];
  const float* proj_w = (const float*)d_in[8];
  const float* proj_b = (const float*)d_in[9];
  const float* ln2_g  = (const float*)d_in[10];
  const float* ln2_b  = (const float*)d_in[11];
  const float* fc_w   = (const float*)d_in[12];
  const float* fc_b   = (const float*)d_in[13];
  const float* mp_w   = (const float*)d_in[14];
  const float* mp_b   = (const float*)d_in[15];
  const float* lnf_g  = (const float*)d_in[16];
  const float* lnf_b  = (const float*)d_in[17];

  // ---- workspace carve (bump allocator, 256B aligned) ----
  char* p = (char*)d_ws;
  auto alloc = [&](size_t bytes) -> char* {
    char* r = p;
    p += (bytes + 255) & ~(size_t)255;
    return r;
  };
  const size_t MN = (size_t)Mrows * Nn;
  float*  x0      = (float*) alloc(MN * 4);
  float*  h       = (float*) alloc(MN * 4);
  float*  x       = (float*) alloc(MN * 4);
  float*  x2      = (float*) alloc(MN * 4);
  __bf16* a_bf    = (__bf16*)alloc(MN * 2);
  __bf16* m_bf    = (__bf16*)alloc(MN * 2);
  __bf16* qkv_bf  = (__bf16*)alloc((size_t)Mrows * N3 * 2);
  __bf16* attn_bf = (__bf16*)alloc(MN * 2);
  __bf16* fc_bf   = (__bf16*)alloc((size_t)Mrows * N4 * 2);
  __bf16* hf_bf   = (__bf16*)alloc(MN * 2);
  __bf16* wteb    = (__bf16*)alloc((size_t)Vv * Nn * 2);
  __bf16* pw_attn = (__bf16*)alloc((size_t)Nn * N3 * 2);
  __bf16* pw_proj = (__bf16*)alloc((size_t)Nn * Nn * 2);
  __bf16* pw_fc   = (__bf16*)alloc((size_t)Nn * N4 * 2);
  __bf16* pw_mp   = (__bf16*)alloc((size_t)N4 * Nn * 2);

  // ---- one-time converts / packs (per launch; deterministic) ----
  {
    long long nw = (long long)Vv * Nn;
    k_f2bf<<<(int)((nw + 255) / 256), 256, 0, stream>>>(wte, wteb, nw);
  }
  auto pack = [&](const float* W, __bf16* P, int K, int Nc) {
    int total = (Nc / 16) * (K / 32) * 512;
    k_pack_w<<<(total + 255) / 256, 256, 0, stream>>>(W, P, K, Nc);
  };
  pack(attn_w, pw_attn, Nn, N3);
  pack(proj_w, pw_proj, Nn, Nn);
  pack(fc_w,   pw_fc,   Nn, N4);
  pack(mp_w,   pw_mp,   N4, Nn);

  // ---- embedding, h = x0 ----
  k_embed<<<(int)(MN / 256), 256, 0, stream>>>(idx, wte, wpe, x0, h);

  auto gemm = [&](const __bf16* A, const __bf16* Bp, const float* bias,
                  const float* resid, float* outf, __bf16* outh, int K, int Nc,
                  int mode) {
    int tiles_n4 = Nc / 64;
    int waves = (Mrows / 16) * tiles_n4;
    k_gemm<<<(waves + 7) / 8, 256, 0, stream>>>(A, Bp, bias, resid, outf, outh,
                                                Mrows, K, Nc, tiles_n4, mode);
  };

  // ---- L looped transformer blocks (shared weights) ----
  for (int l = 0; l < Ll; ++l) {
    // x = h + x0 + se[l]; a = LN1(x)
    k_add_ln<<<Mrows, 256, 0, stream>>>(h, x0, se + (size_t)l * Nn, ln1_g, ln1_b,
                                        x, a_bf);
    // qkv = a @ attn_w + attn_b   (bf16 out)
    gemm(a_bf, pw_attn, attn_b, nullptr, nullptr, qkv_bf, Nn, N3, 0);
    // causal flash attention
    k_attn<<<Bb * Hh * (Tt / 16), 32, 0, stream>>>(qkv_bf, attn_bf);
    // x2 = x + o @ proj_w + proj_b
    gemm(attn_bf, pw_proj, proj_b, x, x2, nullptr, Nn, Nn, 1);
    // m = LN2(x2)
    k_add_ln<<<Mrows, 256, 0, stream>>>(x2, nullptr, nullptr, ln2_g, ln2_b,
                                        nullptr, m_bf);
    // fc = gelu(m @ fc_w + fc_b)  (bf16 out)
    gemm(m_bf, pw_fc, fc_b, nullptr, nullptr, fc_bf, Nn, N4, 2);
    // h = x2 + fc @ mlp_proj_w + mlp_proj_b
    gemm(fc_bf, pw_mp, mp_b, x2, h, nullptr, N4, Nn, 1);
  }

  // ---- final LN + tied lm head on last position ----
  k_add_ln<<<Mrows, 256, 0, stream>>>(h, nullptr, nullptr, lnf_g, lnf_b, nullptr,
                                      hf_bf);
  float* logits = (float*)d_out;
  {
    int VT = (Vv + 15) / 16;
    k_lmhead<<<(VT + 7) / 8, 256, 0, stream>>>(hf_bf, wteb, logits);
  }
  int* steps = (int*)(logits + (size_t)Bb * Vv);
  k_steps<<<(Bb * Tt + 255) / 256, 256, 0, stream>>>(steps);
}
